// RestormerBlock_75436805587648
// MI455X (gfx1250) — matrix operand order repaired and checked by reading, not compiled, not run on hardware
//
#include <hip/hip_runtime.h>
#include <hip/hip_bf16.h>

// ---------------------------------------------------------------------------
// CDNA5 (gfx1250) Restormer block. wave32, WMMA bf16 -> f32 accumulate.
// All matrix fragments are loaded as contiguous 16B (b128) chunks using
// k = 16*(lane/16) + e order for A and B alike (not the ISA table order); TDM stages dwconv planes into LDS.
// ---------------------------------------------------------------------------

typedef __attribute__((ext_vector_type(16))) __bf16 v16bf;
typedef __attribute__((ext_vector_type(8)))  __bf16 bf8v;
typedef __attribute__((ext_vector_type(4)))  __bf16 bf4v;
typedef __attribute__((ext_vector_type(8)))  float  v8f;
typedef __attribute__((ext_vector_type(4)))  float  f4v;

typedef __attribute__((ext_vector_type(4))) unsigned int tdm_u4;
typedef __attribute__((ext_vector_type(8))) int          tdm_i8;
typedef __attribute__((ext_vector_type(4))) int          tdm_i4;

#ifndef __has_builtin
#define __has_builtin(x) 0
#endif
#if __has_builtin(__builtin_amdgcn_tensor_load_to_lds)
#define USE_TDM 1
#else
#define USE_TDM 0
#endif

#define BB      2
#define CDIM    192
#define NHEADS  4
#define DH      48      // per-head channels
#define DHP     64      // padded per-head channels
#define HWN     4096    // 64*64
#define HID     510
#define NQ      16      // query rows per attention block
#define SSTR    (HWN + 8)

union Frag16 { v16bf v; bf8v h[2]; };

static __device__ inline v8f v8f_zero() {
  v8f z;
#pragma unroll
  for (int i = 0; i < 8; ++i) z[i] = 0.f;
  return z;
}

static __device__ inline v8f wmma_bf16(v16bf a, v16bf b, v8f c) {
  return __builtin_amdgcn_wmma_f32_16x16x32_bf16(
      false, a, false, b, (short)0, c, false, false);
}

// A-fragment (16x32, M x K) from row-major K-contiguous storage.
// lanes 0-15: M=lane, K {0..15}; lanes 16-31: M=lane-16, K {16..31}
// -> two contiguous 16B loads per lane.
static __device__ inline v16bf load_a_frag(const __bf16* __restrict__ base, int stride) {
  int lane = threadIdx.x & 31;
  const __bf16* p = base + (lane & 15) * stride + ((lane < 16) ? 0 : 16);
  Frag16 f;
  f.h[0] = *(const bf8v*)(p);
  f.h[1] = *(const bf8v*)(p + 8);
  return f.v;
}

// B-fragment (32x16, K x N) from N-major K-contiguous storage (row = N).
// lane = N (mod 16); element e -> K = (lane<16 ? 0 : 16) + e
// -> one contiguous 32B run per lane (two 16B loads).
static __device__ inline v16bf load_b_frag(const __bf16* __restrict__ base, int stride) {
  int lane = threadIdx.x & 31;
  const __bf16* p = base + (lane & 15) * stride + ((lane < 16) ? 0 : 16);
  Frag16 f;
  f.h[0] = *(const bf8v*)(p);
  f.h[1] = *(const bf8v*)(p + 8);
  return f.v;
}

// ---------------------------------------------------------------------------
// LayerNorm over spatial dims per (b, c)
// ---------------------------------------------------------------------------
__global__ __launch_bounds__(256) void ln2d_kernel(
    const float* __restrict__ x, const float* __restrict__ w,
    const float* __restrict__ b, float* __restrict__ y, int C) {
  int bc = blockIdx.x;
  int c  = bc % C;
  const float* xp = x + (long)bc * HWN;
  float*       yp = y + (long)bc * HWN;
  int tid = threadIdx.x;
  float s = 0.f, s2 = 0.f;
  for (int i = tid; i < HWN; i += 256) { float v = xp[i]; s += v; s2 += v * v; }
  __shared__ float rs[256], rq[256];
  rs[tid] = s; rq[tid] = s2; __syncthreads();
  for (int off = 128; off > 0; off >>= 1) {
    if (tid < off) { rs[tid] += rs[tid + off]; rq[tid] += rq[tid + off]; }
    __syncthreads();
  }
  float mean = rs[0] * (1.f / HWN);
  float var  = rq[0] * (1.f / HWN) - mean * mean;
  float inv  = rsqrtf(var + 1e-6f);
  float sw = w[c], sb = b[c];
  for (int i = tid; i < HWN; i += 256) yp[i] = (xp[i] - mean) * inv * sw + sb;
}

// ---------------------------------------------------------------------------
// conv1x1 GEMM: out(O x P) = W(O x C) * X(C x P) + bias [+ resid]
// 128x128 block, 8 waves (4M x 2N), wave tile 32x64 -> 8 WMMA per K-step.
// As: row-major (K contig). BsT: N-major (K contig) so B-frags are b128 pairs.
// ---------------------------------------------------------------------------
#define GM 128
#define GN 128
#define GK 32

__global__ __launch_bounds__(256) void gemm_bias_kernel(
    const float* __restrict__ X, const float* __restrict__ W,
    const float* __restrict__ bias, const float* __restrict__ resid,
    float* __restrict__ out, int O, int C, int P, long xbs, long obs) {
  __shared__ __bf16 As[GM][GK];   // 8 KB
  __shared__ __bf16 BsT[GN][GK];  // 8 KB
  int bn = blockIdx.x * GN;
  int bm = blockIdx.y * GM;
  const float* Xb = X + (long)blockIdx.z * xbs;
  float*       Ob = out + (long)blockIdx.z * obs;
  const float* Rb = resid ? resid + (long)blockIdx.z * obs : nullptr;

  int tid = threadIdx.x, lane = tid & 31, wave = tid >> 5;
  int wm = wave >> 1, wn = wave & 1;  // 4 x 2 wave grid

  v8f acc[8];
#pragma unroll
  for (int t = 0; t < 8; ++t) acc[t] = v8f_zero();

  for (int k0 = 0; k0 < C; k0 += GK) {
    // stage A: 128x32, guarded scalar loads, packed 8B LDS stores
    for (int i = tid; i < GM * (GK / 4); i += 256) {
      int m = i >> 3, k4 = (i & 7) << 2;
      int gm = bm + m;
      bf4v pk;
#pragma unroll
      for (int j = 0; j < 4; ++j) {
        int gk = k0 + k4 + j;
        pk[j] = (__bf16)((gm < O && gk < C) ? W[(long)gm * C + gk] : 0.f);
      }
      *(bf4v*)&As[m][k4] = pk;
    }
    // stage B transposed: float4 global reads (contig n), scatter to BsT[n][k]
    for (int i = tid; i < GK * (GN / 4); i += 256) {
      int k = i >> 5, n = (i & 31) << 2;
      int gk = k0 + k;
      f4v v;
      if (gk < C) v = *(const f4v*)(Xb + (long)gk * P + bn + n);
      else { v[0] = v[1] = v[2] = v[3] = 0.f; }
#pragma unroll
      for (int j = 0; j < 4; ++j) BsT[n + j][k] = (__bf16)v[j];
    }
    if (k0 + GK < C)
      __builtin_prefetch(Xb + (long)(k0 + GK) * P + bn + tid, 0, 0);
    __syncthreads();

    v16bf a0 = load_a_frag(&As[wm * 32][0], GK);
    v16bf a1 = load_a_frag(&As[wm * 32 + 16][0], GK);
#pragma unroll
    for (int t = 0; t < 4; ++t) {
      v16bf b = load_b_frag(&BsT[wn * 64 + t * 16][0], GK);
      acc[t]     = wmma_bf16(a0, b, acc[t]);
      acc[4 + t] = wmma_bf16(a1, b, acc[4 + t]);
    }
    __syncthreads();
  }

  int hi = (lane >= 16) ? 8 : 0;
  int cb = lane & 15;
#pragma unroll
  for (int h = 0; h < 2; ++h) {
#pragma unroll
    for (int t = 0; t < 4; ++t) {
#pragma unroll
      for (int v = 0; v < 8; ++v) {
        int gm = bm + wm * 32 + h * 16 + v + hi;
        int gn = bn + wn * 64 + t * 16 + cb;
        if (gm < O) {
          long idx = (long)gm * P + gn;
          float r = acc[h * 4 + t][v] + bias[gm];
          if (Rb) r += Rb[idx];
          Ob[idx] = r;
        }
      }
    }
  }
}

// ---------------------------------------------------------------------------
// depthwise 3x3, SAME zero pad; stages the (b,c) 64x64 fp32 plane into LDS
// via the Tensor Data Mover (D# per ISA ch.8) when available.
// ---------------------------------------------------------------------------
__global__ __launch_bounds__(256) void dwconv3x3_kernel(
    const float* __restrict__ in, const float* __restrict__ w,
    const float* __restrict__ bias, float* __restrict__ out, int Ctot) {
  __shared__ float plane[HWN];  // 16 KB
  int bc = blockIdx.x;
  int c  = bc % Ctot;
  const float* ip = in + (long)bc * HWN;

#if USE_TDM
  if (threadIdx.x < 32) {  // one wave issues the DMA
    unsigned lds_addr = (unsigned)(unsigned long long)(void*)plane;  // LDS offset = addr[31:0]
    unsigned long long ga = (unsigned long long)ip;
    tdm_u4 g0;
    g0[0] = 1u;                                                // count=1
    g0[1] = lds_addr;                                          // lds_addr
    g0[2] = (unsigned)(ga & 0xFFFFFFFFu);                      // global_addr[31:0]
    g0[3] = (unsigned)((ga >> 32) & 0x1FFFFFFu) | (2u << 30);  // [56:32] | type=2
    tdm_i8 g1;
    g1[0] = (int)(2u << 16);                   // data_size = 4B
    g1[1] = (int)((HWN & 0xFFFFu) << 16);      // tensor_dim0 lo
    g1[2] = (int)((HWN >> 16) | (1u << 16));   // tensor_dim0 hi | tensor_dim1 lo
    g1[3] = (int)((unsigned)HWN << 16);        // tensor_dim1 hi | tile_dim0 = 4096
    g1[4] = 1;                                 // tile_dim1 = 1
    g1[5] = HWN;                               // tensor_dim0_stride lo
    g1[6] = 0; g1[7] = 0;
    tdm_i4 z4; z4[0] = z4[1] = z4[2] = z4[3] = 0;
#if __has_include(<hip/amd_detail/amd_gfx1250_TDM.h>)
    tdm_i8 z8;
#pragma unroll
    for (int i = 0; i < 8; ++i) z8[i] = 0;
    __builtin_amdgcn_tensor_load_to_lds(g0, g1, z4, z4, z8, 0);
#else
    __builtin_amdgcn_tensor_load_to_lds(g0, g1, z4, z4, 0);
#endif
    __builtin_amdgcn_s_wait_tensorcnt(0);
  }
  __syncthreads();
#else
  for (int i = threadIdx.x; i < HWN; i += 256) plane[i] = ip[i];
  __syncthreads();
#endif

  const float* wp = w + (long)c * 9;  // uniform per block -> scalar loads
  float w00 = wp[0], w01 = wp[1], w02 = wp[2];
  float w10 = wp[3], w11 = wp[4], w12 = wp[5];
  float w20 = wp[6], w21 = wp[7], w22 = wp[8];
  float bi = bias[c];
  float* op = out + (long)bc * HWN;

  for (int i = threadIdx.x; i < HWN; i += 256) {
    int px = i & 63, py = i >> 6;
    float acc = bi;
    bool t = py > 0, bo = py < 63, l = px > 0, r = px < 63;
    if (t) {
      if (l) acc += plane[i - 65] * w00;
      acc += plane[i - 64] * w01;
      if (r) acc += plane[i - 63] * w02;
    }
    if (l) acc += plane[i - 1] * w10;
    acc += plane[i] * w11;
    if (r) acc += plane[i + 1] * w12;
    if (bo) {
      if (l) acc += plane[i + 63] * w20;
      acc += plane[i + 64] * w21;
      if (r) acc += plane[i + 65] * w22;
    }
    op[i] = acc;
  }
}

// ---------------------------------------------------------------------------
// Split q/k/v, L2-normalize q,k over 48 head channels.
//  Qt/Kt: (b, head, pixel, c) bf16, c contiguous, zero-padded to 64
//         -> WMMA frags load as contiguous b128 pairs straight from global.
//  Vn:    (b, head, c, pixel) bf16 (A-matrix of phase C is m-contiguous).
// ---------------------------------------------------------------------------
__global__ __launch_bounds__(256) void qkv_norm_kernel(
    const float* __restrict__ qkv, __bf16* __restrict__ qt,
    __bf16* __restrict__ kt, __bf16* __restrict__ vn, long total) {
  long idx = (long)blockIdx.x * 256 + threadIdx.x;
  if (idx >= total) return;
  int p  = (int)(idx & (HWN - 1));
  long t = idx >> 12;
  int hd = (int)(t % NHEADS);
  int b  = (int)(t / NHEADS);

  const float* base = qkv + (long)b * (3 * CDIM) * HWN;
  const float* qp = base + (long)(hd * DH) * HWN + p;
  const float* kp = base + (long)(CDIM + hd * DH) * HWN + p;
  const float* vp = base + (long)(2 * CDIM + hd * DH) * HWN + p;

  float nq = 0.f, nk = 0.f;
#pragma unroll 4
  for (int c = 0; c < DH; ++c) {
    float q = qp[(long)c * HWN], k = kp[(long)c * HWN];
    nq += q * q; nk += k * k;
  }
  float rq = 1.f / fmaxf(sqrtf(nq), 1e-12f);
  float rk = 1.f / fmaxf(sqrtf(nk), 1e-12f);

  long bh = (long)b * NHEADS + hd;
  __bf16* qo = qt + (bh * HWN + p) * DHP;  // contiguous 64 bf16 (128B aligned)
  __bf16* ko = kt + (bh * HWN + p) * DHP;
#pragma unroll 8
  for (int c = 0; c < DHP; ++c) {
    qo[c] = (__bf16)((c < DH) ? qp[(long)c * HWN] * rq : 0.f);
    ko[c] = (__bf16)((c < DH) ? kp[(long)c * HWN] * rk : 0.f);
  }
  __bf16* vo = vn + bh * DHP * HWN + p;
#pragma unroll 4
  for (int c = 0; c < DH; ++c) vo[(long)c * HWN] = (__bf16)vp[(long)c * HWN];
}

// ---------------------------------------------------------------------------
// Attention over one 16-query tile of one (b, head).
// 16 x 4096 score slab in LDS (~262 KB of the 320 KB WGP LDS).
// ---------------------------------------------------------------------------
__global__ __launch_bounds__(256) void attn_kernel(
    const __bf16* __restrict__ qt, const __bf16* __restrict__ kt,
    const __bf16* __restrict__ vn, const float* __restrict__ temp,
    float* __restrict__ attn_out) {
  extern __shared__ char smem_raw[];
  float* S      = (float*)smem_raw;        // NQ x SSTR
  float* Obuf   = S + NQ * SSTR;           // DH x NQ
  float* red    = Obuf + DH * NQ;          // 256
  float* rowmax = red + 256;               // 16
  float* rowsum = rowmax + 16;             // 16

  int qt_i = blockIdx.x, hd = blockIdx.y, b = blockIdx.z;
  int n0 = qt_i * NQ;
  long bh = (long)b * NHEADS + hd;
  const __bf16* Q = qt + bh * (long)HWN * DHP;  // (pixel, c)
  const __bf16* K = kt + bh * (long)HWN * DHP;  // (pixel, c)
  const __bf16* V = vn + bh * (long)DHP * HWN;  // (c, pixel)
  float tscale = temp[hd];

  int tid = threadIdx.x, lane = tid & 31, wave = tid >> 5;

  // ---- phase A: S = Q K^T ----
  {
    // A = Q rows n0..n0+15, K-dim = c (contiguous) -> b128 frag loads
    v16bf a0 = load_a_frag(Q + (long)n0 * DHP, DHP);
    v16bf a1 = load_a_frag(Q + (long)n0 * DHP + 32, DHP);
    int col = lane & 15;
    int hi  = (lane >= 16) ? 8 : 0;
    for (int mt = wave; mt < HWN / 16; mt += 8) {
      const __bf16* kb = K + (long)mt * 16 * DHP;  // 16 key rows, c contiguous
      v16bf b0 = load_b_frag(kb, DHP);             // c 0..31
      v16bf b1 = load_b_frag(kb + 32, DHP);        // c 32..63 (zero padded)
      v8f s = v8f_zero();
      s = wmma_bf16(a0, b0, s);
      s = wmma_bf16(a1, b1, s);
#pragma unroll
      for (int v = 0; v < 8; ++v) S[(v + hi) * SSTR + mt * 16 + col] = s[v];
    }
  }
  __syncthreads();

  // ---- phase B: softmax in LDS (16 threads per query row) ----
  {
    int r = tid >> 4, cs = tid & 15;
    float mx = -3.4e38f;
    for (int m = cs; m < HWN; m += 16) {
      float v = S[r * SSTR + m] * tscale;
      S[r * SSTR + m] = v;
      mx = fmaxf(mx, v);
    }
    red[tid] = mx; __syncthreads();
    if (cs == 0) {
      float m2 = red[tid];
#pragma unroll
      for (int i = 1; i < 16; ++i) m2 = fmaxf(m2, red[tid + i]);
      rowmax[r] = m2;
    }
    __syncthreads();
    float rm = rowmax[r], sm = 0.f;
    for (int m = cs; m < HWN; m += 16) {
      float e = __expf(S[r * SSTR + m] - rm);
      S[r * SSTR + m] = e;
      sm += e;
    }
    red[tid] = sm; __syncthreads();
    if (cs == 0) {
      float s2 = 0.f;
#pragma unroll
      for (int i = 0; i < 16; ++i) s2 += red[tid + i];
      rowsum[r] = s2;
    }
  }
  __syncthreads();

  for (int i = tid; i < DH * NQ; i += 256) Obuf[i] = 0.f;
  __syncthreads();

  // ---- phase C: O = V P^T (K = 4096 split across 8 waves) ----
  {
    v8f acc[3];
#pragma unroll
    for (int t = 0; t < 3; ++t) acc[t] = v8f_zero();
    int col = lane & 15;
    int kb2 = (lane < 16) ? 0 : 16;
    for (int m0 = wave * 32; m0 < HWN; m0 += 256) {
      // B[k=m][n] = P[n][m]: 16 contiguous fp32 from LDS -> 4x ds_load_b128
      const float* sp = &S[col * SSTR + m0 + kb2];
      f4v s0 = *(const f4v*)(sp);
      f4v s1 = *(const f4v*)(sp + 4);
      f4v s2 = *(const f4v*)(sp + 8);
      f4v s3 = *(const f4v*)(sp + 12);
      v16bf bf;
#pragma unroll
      for (int j = 0; j < 4; ++j) {
        bf[j]      = (__bf16)s0[j];
        bf[4 + j]  = (__bf16)s1[j];
        bf[8 + j]  = (__bf16)s2[j];
        bf[12 + j] = (__bf16)s3[j];
      }
#pragma unroll
      for (int ct = 0; ct < 3; ++ct) {
        // A[c][m], m contiguous -> b128 pair from global
        v16bf af = load_a_frag(V + (long)(ct * 16) * HWN + m0, HWN);
        acc[ct] = wmma_bf16(af, bf, acc[ct]);
      }
    }
    int hi = (lane >= 16) ? 8 : 0;
#pragma unroll
    for (int ct = 0; ct < 3; ++ct)
#pragma unroll
      for (int v = 0; v < 8; ++v)
        atomicAdd(&Obuf[(ct * 16 + v + hi) * NQ + col], acc[ct][v]);  // ds_add_f32
  }
  __syncthreads();

  for (int i = tid; i < DH * NQ; i += 256) {
    int c = i / NQ, n = i % NQ;
    attn_out[((long)b * CDIM + hd * DH + c) * HWN + n0 + n] = Obuf[i] / rowsum[n];
  }
}

// ---------------------------------------------------------------------------
// GDFN gate: out[c] = gelu_exact(y[c]) * y[510+c]
// ---------------------------------------------------------------------------
__global__ __launch_bounds__(256) void gelu_gate_kernel(
    const float* __restrict__ yin, float* __restrict__ out, long total) {
  long idx = (long)blockIdx.x * 256 + threadIdx.x;
  if (idx >= total) return;
  int p  = (int)(idx & (HWN - 1));
  long t = idx >> 12;
  int c  = (int)(t % HID);
  int b  = (int)(t / HID);
  long base = (long)b * (2 * HID) * HWN;
  float a = yin[base + (long)c * HWN + p];
  float g = yin[base + (long)(HID + c) * HWN + p];
  float ge = 0.5f * a * (1.f + erff(a * 0.70710678118654752f));
  out[idx] = ge * g;
}

// ---------------------------------------------------------------------------
// Host orchestration
// ---------------------------------------------------------------------------
extern "C" void kernel_launch(void* const* d_in, const int* in_sizes, int n_in,
                              void* d_out, int out_size, void* d_ws, size_t ws_size,
                              hipStream_t stream) {
  (void)in_sizes; (void)n_in; (void)out_size; (void)ws_size;
  const float* x       = (const float*)d_in[0];
  const float* ln1_w   = (const float*)d_in[1];
  const float* ln1_b   = (const float*)d_in[2];
  const float* qkv_w   = (const float*)d_in[3];
  const float* qkv_b   = (const float*)d_in[4];
  const float* qkv_dww = (const float*)d_in[5];
  const float* qkv_dwb = (const float*)d_in[6];
  const float* temp    = (const float*)d_in[7];
  const float* proj_w  = (const float*)d_in[8];
  const float* proj_b  = (const float*)d_in[9];
  const float* ln2_w   = (const float*)d_in[10];
  const float* ln2_b   = (const float*)d_in[11];
  const float* pin_w   = (const float*)d_in[12];
  const float* pin_b   = (const float*)d_in[13];
  const float* gdw_w   = (const float*)d_in[14];
  const float* gdw_b   = (const float*)d_in[15];
  const float* pout_w  = (const float*)d_in[16];
  const float* pout_b  = (const float*)d_in[17];
  float* out = (float*)d_out;

  char* ws = (char*)d_ws;
  size_t off = 0;
  auto carve = [&](size_t bytes) { void* p = ws + off; off += (bytes + 255) & ~(size_t)255; return p; };
  float*  big0     = (float*)carve((size_t)BB * (2 * HID) * HWN * 4);
  float*  big1     = (float*)carve((size_t)BB * (2 * HID) * HWN * 4);
  float*  ybuf     = (float*)carve((size_t)BB * CDIM * HWN * 4);
  float*  attn_out = (float*)carve((size_t)BB * CDIM * HWN * 4);
  float*  gated    = (float*)carve((size_t)BB * HID * HWN * 4);
  __bf16* qt_buf   = (__bf16*)carve((size_t)BB * NHEADS * HWN * DHP * 2);
  __bf16* kt_buf   = (__bf16*)carve((size_t)BB * NHEADS * HWN * DHP * 2);
  __bf16* vn_buf   = (__bf16*)carve((size_t)BB * NHEADS * DHP * HWN * 2);

  const long mid_s = (long)CDIM * HWN;
  const long qkv_s = (long)(3 * CDIM) * HWN;
  const long pin_s = (long)(2 * HID) * HWN;
  const long gat_s = (long)HID * HWN;

  // 1) LN1
  ln2d_kernel<<<BB * CDIM, 256, 0, stream>>>(x, ln1_w, ln1_b, ybuf, CDIM);
  // 2) qkv conv1x1
  gemm_bias_kernel<<<dim3(HWN / GN, (3 * CDIM + GM - 1) / GM, BB), 256, 0, stream>>>(
      ybuf, qkv_w, qkv_b, nullptr, big0, 3 * CDIM, CDIM, HWN, mid_s, qkv_s);
  // 3) qkv depthwise 3x3 (TDM-staged planes)
  dwconv3x3_kernel<<<BB * 3 * CDIM, 256, 0, stream>>>(big0, qkv_dww, qkv_dwb, big1, 3 * CDIM);
  // 4) split + L2 norm + bf16 pack (transposed Q/K)
  {
    long total = (long)BB * NHEADS * HWN;
    qkv_norm_kernel<<<(total + 255) / 256, 256, 0, stream>>>(big1, qt_buf, kt_buf, vn_buf, total);
  }
  // 5) attention
  {
    size_t smem = (size_t)NQ * SSTR * 4 + (size_t)DH * NQ * 4 + 256 * 4 + 32 * 4;
    attn_kernel<<<dim3(HWN / NQ, NHEADS, BB), 256, smem, stream>>>(qt_buf, kt_buf, vn_buf,
                                                                   temp, attn_out);
  }
  // 6) proj conv1x1 + residual(x) -> d_out = x1
  gemm_bias_kernel<<<dim3(HWN / GN, (CDIM + GM - 1) / GM, BB), 256, 0, stream>>>(
      attn_out, proj_w, proj_b, x, out, CDIM, CDIM, HWN, mid_s, mid_s);
  // 7) LN2
  ln2d_kernel<<<BB * CDIM, 256, 0, stream>>>(out, ln2_w, ln2_b, ybuf, CDIM);
  // 8) pin conv1x1 (M tail guarded)
  gemm_bias_kernel<<<dim3(HWN / GN, (2 * HID + GM - 1) / GM, BB), 256, 0, stream>>>(
      ybuf, pin_w, pin_b, nullptr, big0, 2 * HID, CDIM, HWN, mid_s, pin_s);
  // 9) gdfn depthwise 3x3
  dwconv3x3_kernel<<<BB * 2 * HID, 256, 0, stream>>>(big0, gdw_w, gdw_b, big1, 2 * HID);
  // 10) gelu gate
  {
    long total = (long)BB * HID * HWN;
    gelu_gate_kernel<<<(total + 255) / 256, 256, 0, stream>>>(big1, gated, total);
  }
  // 11) pout conv1x1 (K=510 tail guarded) + residual(x1)
  gemm_bias_kernel<<<dim3(HWN / GN, (CDIM + GM - 1) / GM, BB), 256, 0, stream>>>(
      gated, pout_w, pout_b, out, out, CDIM, HID, HWN, gat_s, mid_s);
}
